// StageCCriterion_44143673869067
// MI455X (gfx1250) — matrix operand order repaired and checked by reading, not compiled, not run on hardware
//
#include <hip/hip_runtime.h>
#include <hip/hip_bf16.h>
#include <climits>

// ---------------------------------------------------------------------------
// StageCCriterion on MI455X (gfx1250, wave32).
// Losses are memory-bound reductions except the camera-pair consistency
// einsum, recast as 240 dense 256x256x256 GEMMs on the WMMA pipe
// (v_wmma_f32_16x16x32_f16, f32 accumulate), with the B operand staged in
// LDS as f16 (column-major, padded stride), the 0/1 A operand generated
// branch-free from id arrays, and B tiles software-pipelined 2-deep so the
// per-WMMA LDS waits relax from dscnt==0 to dscnt<=2.
// ---------------------------------------------------------------------------

namespace {
constexpr int kB = 16, kC = 6, kQ = 256, kT = 256, kD = 256;
constexpr int kBCQ   = kB * kC * kQ;          // 24576
constexpr int kPairs = (kC * (kC - 1)) / 2;   // 15
}

typedef __attribute__((ext_vector_type(16))) _Float16 v16h;
typedef __attribute__((ext_vector_type(8)))  _Float16 v8h;
typedef __attribute__((ext_vector_type(8)))  float    v8f;

// acc layout (floats in d_ws):
// [0] sum det^2        [1] entropy numerator  [2] score numerator
// [3] cam count        [4] nll numerator      [5] valid count
// [6] box numerator    [7] consistency num    [8] consistency count

__global__ void init_acc_kernel(float* __restrict__ acc) {
    if (threadIdx.x < 16) acc[threadIdx.x] = 0.0f;
}

// ---------------------------------------------------------------------------
// Stable partition per (b,c): targets first (original order preserved).
// ---------------------------------------------------------------------------
__global__ void pack_kernel(const unsigned char* __restrict__ target_mask,
                            const unsigned char* __restrict__ cam_mask,
                            const int* __restrict__ ids,
                            const float* __restrict__ boxes,
                            const int* __restrict__ img_h_p,
                            const int* __restrict__ img_w_p,
                            int* __restrict__ qids,
                            int* __restrict__ slots,
                            int* __restrict__ qvalid,
                            float* __restrict__ boxt) {
    __shared__ int s[kQ];
    __shared__ int orderS[kQ];
    const int bc  = blockIdx.x;
    const int tid = threadIdx.x;

    const int m = target_mask[bc * kQ + tid] ? 1 : 0;
    s[tid] = m;
    __syncthreads();
    // Hillis-Steele inclusive scan over 256 elements.
    for (int off = 1; off < kQ; off <<= 1) {
        int v = (tid >= off) ? s[tid - off] : 0;
        __syncthreads();
        s[tid] += v;
        __syncthreads();
    }
    const int incl  = s[tid];
    const int excl  = incl - m;
    const int total = s[kQ - 1];
    const int rank  = m ? excl : (total + (tid - excl));
    orderS[rank] = tid;
    __syncthreads();

    const bool cam   = cam_mask[bc] != 0;
    const float iw   = (float)img_w_p[0];
    const float ih   = (float)img_h_p[0];
    const int   src  = orderS[tid];
    const bool valid = (tid < total) && cam;
    const int  sid   = ids[bc * kQ + src];
    const int  qid   = valid ? sid : -1;

    qids  [bc * kQ + tid] = qid;
    qvalid[bc * kQ + tid] = valid ? 1 : 0;
    const int mq = (qid >= 0) ? qid : 0;   // max(qid, 0)
    slots[bc * kQ + tid] = mq % kT;

#pragma unroll
    for (int k = 0; k < 4; ++k) {
        const float nrm = (k & 1) ? ih : iw;
        float bg = boxes[((size_t)bc * kQ + src) * 4 + k] / nrm;
        bg = fminf(fmaxf(bg, 0.0f), 1.0f);
        boxt[((size_t)bc * kQ + tid) * 4 + k] = valid ? bg : 0.0f;
    }
}

// ---------------------------------------------------------------------------
// Fused flat reductions (float4): det_norm numerator + entropy numerator.
// det_tokens and assoc are both (B,C,Q,256) -> same flat extent; every
// float4 stays inside a single (b,c), so one cam_mask lookup per vector.
// ---------------------------------------------------------------------------
__global__ void flat_kernel(const float* __restrict__ det_tokens,
                            const float* __restrict__ assoc,
                            const unsigned char* __restrict__ cam_mask,
                            float* __restrict__ acc) {
    const int N4 = (kB * kC * kQ * kT) / 4;   // 1572864
    const float4* dt4 = (const float4*)det_tokens;
    const float4* as4 = (const float4*)assoc;
    float sq = 0.0f, ent = 0.0f;
    for (int i = blockIdx.x * blockDim.x + threadIdx.x; i < N4;
         i += gridDim.x * blockDim.x) {
        const float4 x = dt4[i];
        sq += x.x * x.x + x.y * x.y + x.z * x.z + x.w * x.w;
        const int bc = i >> 14;               // (i*4) / 65536
        if (cam_mask[bc]) {
            const float4 a = as4[i];
            ent -= logf(fmaxf(a.x, 1e-8f)) * a.x;
            ent -= logf(fmaxf(a.y, 1e-8f)) * a.y;
            ent -= logf(fmaxf(a.z, 1e-8f)) * a.z;
            ent -= logf(fmaxf(a.w, 1e-8f)) * a.w;
        }
    }
    __shared__ float r0[256];
    __shared__ float r1[256];
    r0[threadIdx.x] = sq;
    r1[threadIdx.x] = ent;
    __syncthreads();
    for (int off = 128; off >= 1; off >>= 1) {
        if ((int)threadIdx.x < off) {
            r0[threadIdx.x] += r0[threadIdx.x + off];
            r1[threadIdx.x] += r1[threadIdx.x + off];
        }
        __syncthreads();
    }
    if (threadIdx.x == 0) {
        atomicAdd(&acc[0], r0[0]);
        atomicAdd(&acc[1], r1[0]);
    }
}

// ---------------------------------------------------------------------------
// Per-(b,c,q): BCE score loss, NLL pair loss, L1 box loss + mask counts.
// ---------------------------------------------------------------------------
__global__ void perq_kernel(const float* __restrict__ det_scores,
                            const float* __restrict__ det_boxes,
                            const float* __restrict__ assoc,
                            const unsigned char* __restrict__ cam_mask,
                            const int* __restrict__ slots,
                            const int* __restrict__ qvalid,
                            const float* __restrict__ boxt,
                            float* __restrict__ acc) {
    const int g  = blockIdx.x * 256 + threadIdx.x;   // < 24576, exact grid
    const int bc = g >> 8;
    const bool cam = cam_mask[bc] != 0;
    const int  vld = qvalid[g];

    const float t = vld ? 1.0f : 0.0f;
    const float p = fminf(fmaxf(det_scores[g], 1e-6f), 1.0f - 1e-6f);
    const float bce = -(t * logf(p) + (1.0f - t) * log1pf(-p));

    const int   slot = slots[g];
    const float av   = assoc[(size_t)g * kT + slot];
    const float nll  = -logf(fmaxf(av, 1e-8f));

    float bl = 0.0f;
#pragma unroll
    for (int k = 0; k < 4; ++k)
        bl += fabsf(det_boxes[(size_t)g * 4 + k] - boxt[(size_t)g * 4 + k]);

    __shared__ float r[5][256];
    r[0][threadIdx.x] = cam ? bce : 0.0f;
    r[1][threadIdx.x] = cam ? 1.0f : 0.0f;
    r[2][threadIdx.x] = vld ? nll : 0.0f;
    r[3][threadIdx.x] = vld ? 1.0f : 0.0f;
    r[4][threadIdx.x] = vld ? bl  : 0.0f;
    __syncthreads();
    for (int off = 128; off >= 1; off >>= 1) {
        if ((int)threadIdx.x < off) {
#pragma unroll
            for (int j = 0; j < 5; ++j)
                r[j][threadIdx.x] += r[j][threadIdx.x + off];
        }
        __syncthreads();
    }
    if (threadIdx.x == 0) {
        atomicAdd(&acc[2], r[0][0]);
        atomicAdd(&acc[3], r[1][0]);
        atomicAdd(&acc[4], r[2][0]);
        atomicAdd(&acc[5], r[3][0]);
        atomicAdd(&acc[6], r[4][0]);
    }
}

// ---------------------------------------------------------------------------
// Consistency term. One block per (b, c<d) pair (240 blocks, 8 waves).
// peer = M @ assoc_d.  assoc_d is staged in LDS (f16, column-major, padded
// stride) in 64-row K-blocks; each wave owns two 16-row q-strips and keeps
// 16 t-tile accumulators live across the K loop.  A is generated branch-
// free from id arrays via a non-matching sentinel (no exec divergence).
// B tiles are software-pipelined one tile ahead of the WMMA that consumes
// them so LDS latency overlaps matrix issue.
// ---------------------------------------------------------------------------
__global__ void __launch_bounds__(256)
consistency_kernel(const float* __restrict__ assoc,
                   const int* __restrict__ qids,
                   float* __restrict__ acc) {
    constexpr int KBLK = 64;   // K rows staged per block iteration
    constexpr int STR  = 72;   // padded per-column row stride in f16 elems

    __shared__ int      s_qc[kQ];
    __shared__ int      s_qd[kQ];
    __shared__ int      s_cnt[kQ];
    __shared__ float    s_term[kQ];
    __shared__ _Float16 s_bt[kT * STR];   // 36864 B, column-major f16
    __shared__ float    s_red[kQ];
    __shared__ float    s_redc[kQ];

    const int tid    = threadIdx.x;
    const int lane   = tid & 31;
    const int wave   = tid >> 5;
    const int halfId = lane >> 4;    // 0: lanes 0-15, 1: lanes 16-31
    const int idx    = lane & 15;

    // Decode (b, c, d) with c < d from blockIdx.
    const int pi = blockIdx.x % kPairs;
    const int b  = blockIdx.x / kPairs;
    int c = 0, rem = pi;
    while (rem >= (kC - 1 - c)) { rem -= (kC - 1 - c); ++c; }
    const int d = c + 1 + rem;

    const float* assoc_c = assoc + (size_t)(b * kC + c) * kQ * kT;
    const float* assoc_d = assoc + (size_t)(b * kC + d) * kQ * kT;

    s_qc[tid] = qids[(b * kC + c) * kQ + tid];
    s_qd[tid] = qids[(b * kC + d) * kQ + tid];
    __syncthreads();

    // cnt[q] = number of matching valid p in camera d (branch-free scan).
    {
        const int qa = s_qc[tid];
        const int qs = (qa >= 0) ? qa : INT_MIN;
        int cn = 0;
#pragma unroll 8
        for (int p = 0; p < kQ; ++p) cn += (s_qd[p] == qs) ? 1 : 0;
        s_cnt[tid] = cn;
    }

    const int4* qd4 = (const int4*)s_qd;

    for (int qt = 0; qt < 2; ++qt) {
        const int Q0 = (wave * 2 + qt) * 16;   // this wave's q-strip
        const int qa = s_qc[Q0 + idx];
        const int qsent = (qa >= 0) ? qa : INT_MIN;   // never matches

        v8f accv[16];
#pragma unroll
        for (int tt = 0; tt < 16; ++tt) accv[tt] = v8f{};

        for (int kb = 0; kb < 4; ++kb) {
            __syncthreads();   // previous users of s_bt are done
            // Stage 64x256 f32 -> f16, column-major: s_bt[col*STR + row].
#pragma unroll 8
            for (int r = 0; r < KBLK; ++r)
                s_bt[tid * STR + r] =
                    (_Float16)assoc_d[(size_t)(kb * KBLK + r) * kT + tid];
            __syncthreads();

            for (int ks = 0; ks < 2; ++ks) {
                // --- A (16x32 f16): row M = idx, interleaved K (ISA 7.12.2).
                // K indices are two runs of 8 consecutive values.
                const int base0 = kb * KBLK + ks * 32 + halfId * 8;
                const int4 w0 = qd4[(base0 >> 2) + 0];
                const int4 w1 = qd4[(base0 >> 2) + 1];
                const int4 w2 = qd4[((base0 + 16) >> 2) + 0];
                const int4 w3 = qd4[((base0 + 16) >> 2) + 1];
                const _Float16 one = (_Float16)1.0f, zero = (_Float16)0.0f;
                v16h a;
                a[0]  = (w0.x == qsent) ? one : zero;
                a[1]  = (w0.y == qsent) ? one : zero;
                a[2]  = (w0.z == qsent) ? one : zero;
                a[3]  = (w0.w == qsent) ? one : zero;
                a[4]  = (w1.x == qsent) ? one : zero;
                a[5]  = (w1.y == qsent) ? one : zero;
                a[6]  = (w1.z == qsent) ? one : zero;
                a[7]  = (w1.w == qsent) ? one : zero;
                a[8]  = (w2.x == qsent) ? one : zero;
                a[9]  = (w2.y == qsent) ? one : zero;
                a[10] = (w2.z == qsent) ? one : zero;
                a[11] = (w2.w == qsent) ? one : zero;
                a[12] = (w3.x == qsent) ? one : zero;
                a[13] = (w3.y == qsent) ? one : zero;
                a[14] = (w3.z == qsent) ? one : zero;
                a[15] = (w3.w == qsent) ? one : zero;

                const int rowOff = ks * 32 + halfId * 8;
                // B (32x16 f16): col N = idx; interleaved K matching A
                // (two runs of 8 consecutive K per lane half).  Software-
                // pipelined one t-tile ahead of the consuming WMMA.
                auto loadB = [&](int tt) -> v16h {
                    const v8h* bp = (const v8h*)
                        &s_bt[(tt * 16 + idx) * STR + rowOff];
                    const v8h blo = bp[0];
                    const v8h bhi = bp[2];
                    v16h bm;
#pragma unroll
                    for (int i = 0; i < 8; ++i) {
                        bm[i]     = blo[i];
                        bm[i + 8] = bhi[i];
                    }
                    return bm;
                };

                v16h bcur = loadB(0);
#pragma unroll
                for (int tt = 0; tt < 16; ++tt) {
                    v16h bnext;
                    if (tt < 15) bnext = loadB(tt + 1);
                    accv[tt] = __builtin_amdgcn_wmma_f32_16x16x32_f16(
                        false, a, false, bcur, (short)0, accv[tt],
                        false, false);
                    bcur = bnext;
                }
            }
        }

        // Epilogue: C/D layout -> VGPR r holds row (halfId*8 + r), col idx.
        // Accumulate sum_t diff^2 per q-row in registers across t-tiles.
        float tacc[8];
#pragma unroll
        for (int r = 0; r < 8; ++r) tacc[r] = 0.0f;
#pragma unroll
        for (int tt = 0; tt < 16; ++tt) {
#pragma unroll
            for (int r = 0; r < 8; ++r) {
                const int   q    = Q0 + halfId * 8 + r;
                const float cq   = (float)s_cnt[q];
                const float peer = accv[tt][r] / fmaxf(cq, 1.0f);
                const float ref  = assoc_c[(size_t)q * kT + tt * 16 + idx];
                const float df   = ref - peer;
                tacc[r] += df * df;
            }
        }
#pragma unroll
        for (int r = 0; r < 8; ++r) {
            float v = tacc[r];
            v += __shfl_xor(v, 8, 16);
            v += __shfl_xor(v, 4, 16);
            v += __shfl_xor(v, 2, 16);
            v += __shfl_xor(v, 1, 16);
            if (idx == 0) s_term[Q0 + halfId * 8 + r] = v;  // exclusive owner
        }
    }
    __syncthreads();

    // Masked mean contribution: term[q]/T where cnt[q] > 0.
    float tsum = 0.0f, tcnt = 0.0f;
    if (s_cnt[tid] > 0) {
        tsum = s_term[tid] * (1.0f / (float)kT);
        tcnt = 1.0f;
    }
    s_red[tid]  = tsum;
    s_redc[tid] = tcnt;
    __syncthreads();
    for (int off = 128; off >= 1; off >>= 1) {
        if (tid < off) {
            s_red[tid]  += s_red[tid + off];
            s_redc[tid] += s_redc[tid + off];
        }
        __syncthreads();
    }
    if (tid == 0) {
        atomicAdd(&acc[7], s_red[0]);
        atomicAdd(&acc[8], s_redc[0]);
    }
}

// ---------------------------------------------------------------------------
// Combine accumulators into the six output scalars.
// ---------------------------------------------------------------------------
__global__ void finalize_kernel(const float* __restrict__ acc,
                                float* __restrict__ out) {
    if (threadIdx.x != 0 || blockIdx.x != 0) return;
    const float detNorm = acc[0] / (float)((double)kB * kC * kQ * kD);
    const float camCnt  = fmaxf(acc[3], 1.0f);
    const float ent     = acc[1] / camCnt;
    const float score   = acc[2] / camCnt;
    const float vCnt    = acc[5];
    const float pair    = acc[4] / fmaxf(vCnt, 1.0f);
    const float box     = acc[6] / fmaxf(4.0f * vCnt, 1.0f);
    const float cons    = acc[7] / fmaxf(acc[8], 1.0f);
    const float detSup  = score + box;
    const float total   = detNorm + detSup + ent + pair + cons;
    out[0] = total;
    out[1] = detNorm;
    out[2] = detSup;
    out[3] = ent;
    out[4] = pair;
    out[5] = cons;
}

// ---------------------------------------------------------------------------
extern "C" void kernel_launch(void* const* d_in, const int* in_sizes, int n_in,
                              void* d_out, int out_size, void* d_ws,
                              size_t ws_size, hipStream_t stream) {
    (void)in_sizes; (void)n_in; (void)out_size; (void)ws_size;

    const float*         det_tokens  = (const float*)d_in[0];
    const float*         det_scores  = (const float*)d_in[1];
    const float*         det_boxes   = (const float*)d_in[2];
    const float*         assoc       = (const float*)d_in[3];
    const float*         boxes       = (const float*)d_in[4];
    const unsigned char* cam_mask    = (const unsigned char*)d_in[5];
    const unsigned char* target_mask = (const unsigned char*)d_in[6];
    const int*           ids         = (const int*)d_in[7];
    const int*           img_h       = (const int*)d_in[8];
    const int*           img_w       = (const int*)d_in[9];

    // Workspace layout.
    float* acc    = (float*)d_ws;                      // 16 floats
    int*   qids   = (int*)((char*)d_ws + 256);
    int*   slots  = qids + kBCQ;
    int*   qvalid = slots + kBCQ;
    float* boxt   = (float*)(qvalid + kBCQ);           // kBCQ*4 floats

    hipLaunchKernelGGL(init_acc_kernel, dim3(1), dim3(32), 0, stream, acc);
    hipLaunchKernelGGL(pack_kernel, dim3(kB * kC), dim3(kQ), 0, stream,
                       target_mask, cam_mask, ids, boxes, img_h, img_w,
                       qids, slots, qvalid, boxt);
    hipLaunchKernelGGL(flat_kernel, dim3(1024), dim3(256), 0, stream,
                       det_tokens, assoc, cam_mask, acc);
    hipLaunchKernelGGL(perq_kernel, dim3(kBCQ / 256), dim3(256), 0, stream,
                       det_scores, det_boxes, assoc, cam_mask,
                       slots, qvalid, boxt, acc);
    hipLaunchKernelGGL(consistency_kernel, dim3(kB * kPairs), dim3(256), 0,
                       stream, assoc, qids, acc);
    hipLaunchKernelGGL(finalize_kernel, dim3(1), dim3(1), 0, stream,
                       acc, (float*)d_out);
}
